// SuctionNet_prob_4741643895570
// MI455X (gfx1250) — matrix operand order repaired and checked by reading, not compiled, not run on hardware
//
#include <hip/hip_runtime.h>
#include <math.h>

// ---------------------------------------------------------------------------
// SuctionNet: sparse conv (3->64) -> sparse conv (64->256) -> linear heads.
// Conv2 recast as gathered-A GEMM (M=200000, K=1728, N=256) on the bf16 WMMA
// path with f32 accumulation; heads fused into the GEMM epilogue so h2 is
// never materialized.  A-tiles are double-buffered in LDS and staged with
// CDNA5 async-to-LDS transfers (ASYNCcnt) so the neighbor gather for tile k+1
// overlaps the 16 WMMAs of tile k.
// ---------------------------------------------------------------------------

typedef __attribute__((ext_vector_type(16))) __bf16       v16bf;
typedef __attribute__((ext_vector_type(8)))  float        v8f;
typedef __attribute__((ext_vector_type(8)))  unsigned int v8u;

#define NVOX  200000
#define KNBR  27
#define C1    64
#define COUT  256
#define KTOT  (KNBR * C1)        // 1728
#define ROWS  72                  // LDS row stride in halves (64 + 8 pad)

#define USE_ASYNC_LDS 1

__device__ __forceinline__ unsigned short f32_to_bf16_rne(float f) {
  unsigned int x = __float_as_uint(f);
  unsigned int r = x + 0x7FFFu + ((x >> 16) & 1u);
  return (unsigned short)(r >> 16);
}

#if USE_ASYNC_LDS
// Copy 32 contiguous bytes global -> LDS via two async b128 transfers.
// INST_OFFSET is added to both the global and the LDS address, so a single
// (lds, global) address pair covers both halves.
__device__ __forceinline__ void async_copy32(unsigned short* lds_dst,
                                             const unsigned short* gsrc) {
  unsigned int       ldsoff = (unsigned int)(uintptr_t)lds_dst;       // addr[31:0]
  unsigned long long gaddr  = (unsigned long long)(uintptr_t)gsrc;
  asm volatile("global_load_async_to_lds_b128 %0, %1, off\n\t"
               "global_load_async_to_lds_b128 %0, %1, off offset:16"
               :: "v"(ldsoff), "v"(gaddr) : "memory");
}
__device__ __forceinline__ void async_wait0() {
  asm volatile("s_wait_asynccnt 0x0" ::: "memory");
}
#endif

// -------------------------- conv1: [N,3] -> [N,64] bf16 ---------------------
__global__ void sn_conv1_kernel(const float* __restrict__ feats,
                                const int*   __restrict__ nidx1,
                                const float* __restrict__ W1,   // [27,3,64]
                                const float* __restrict__ b1,   // [64]
                                unsigned short* __restrict__ h1bf) {
  int tid = threadIdx.x;
  int n = blockIdx.x * 4 + (tid >> 6);   // 4 voxels per 256-thread block
  int c = tid & 63;
  if (n >= NVOX) return;
  float acc = b1[c];
  const int* nrow = nidx1 + n * KNBR;
  #pragma unroll 3
  for (int k = 0; k < KNBR; ++k) {
    int idx = nrow[k];
    const float* f = feats + idx * 3;
    const float* w = W1 + (k * 3) * 64 + c;
    acc = fmaf(f[0], w[0],   acc);
    acc = fmaf(f[1], w[64],  acc);
    acc = fmaf(f[2], w[128], acc);
  }
  acc = fmaxf(acc, 0.0f);
  h1bf[n * 64 + c] = f32_to_bf16_rne(acc);
}

// ---------------- W2 repack into WMMA B-fragment-major bf16 ----------------
// Fragment tile (kb,nb): 32 K x 16 N.  Storage: tile-major, then lane, then
// the lane's 16 half elements contiguous (32B per lane -> coalesced loads).
// B layout (16-bit 32x16): lanes 0-15 hold K=0..7,16..23 (K=e+8*(e/8)), lanes 16-31 hold
// K=8..15,24..31; column = lane%16.
__global__ void sn_w2swz_kernel(const float* __restrict__ W2,   // [1728,256]
                                unsigned short* __restrict__ w2swz) {
  int idx = blockIdx.x * blockDim.x + threadIdx.x;
  if (idx >= KTOT * COUT) return;
  int e    = idx & 15;
  int lane = (idx >> 4) & 31;
  int tile = idx >> 9;          // kb*16 + nb
  int kb   = tile >> 4;
  int nb   = tile & 15;
  int K    = e + ((e >> 3) << 3) + ((lane >> 4) << 3);
  int col  = nb * 16 + (lane & 15);
  w2swz[idx] = f32_to_bf16_rne(W2[(kb * 32 + K) * COUT + col]);
}

// ------------- conv2 GEMM + fused heads (per-voxel reductions) -------------
// Block: 256 threads = 8 waves; 64 rows x 256 cols per block.
// Wave (w_m,w_n): rows w_m*16..+16, cols w_n*128..+128 (8 WMMA C tiles).
__global__ __launch_bounds__(256, 2)
void sn_conv2_gemm_kernel(const unsigned short* __restrict__ h1bf,
                          const int*            __restrict__ nidx2,
                          const unsigned short* __restrict__ w2swz,
                          const float*          __restrict__ b2,
                          const float*          __restrict__ score_w,
                          const float*          __restrict__ sigma_w,
                          float* __restrict__ s_vox,
                          float* __restrict__ g_vox) {
  __shared__ unsigned short Alds[2][64 * ROWS];
  __shared__ float red_s[128];
  __shared__ float red_g[128];

  const int tid  = threadIdx.x;
  const int lane = tid & 31;
  const int w    = tid >> 5;
  const int w_m  = w & 3;
  const int w_n  = w >> 2;
  const int m0   = blockIdx.x * 64;

  v8f acc[8];
  #pragma unroll
  for (int t = 0; t < 8; ++t) acc[t] = (v8f){0.f,0.f,0.f,0.f,0.f,0.f,0.f,0.f};

  // gather assignment: 4 threads per row, 32B (16 halves) each
  const int grow = tid >> 2;                 // 0..63
  const int gq   = tid & 3;
  const int* nrow = nidx2 + (m0 + grow) * KNBR;
  const int ldoff = grow * ROWS + gq * 16;   // halves within a buffer

  // A-fragment addressing (ISA 16-bit A 16x32 layout):
  //   lane -> M row = lane%16 ; K pair j -> (j<4?0:16) + (j&3)*2 + 8*(lane/16)
  const int arow = w_m * 16 + (lane & 15);
  const int ahi  = (lane >> 4) << 3;

  // prologue: stage tile k=0 into buffer 0
#if USE_ASYNC_LDS
  async_copy32(&Alds[0][ldoff], h1bf + (size_t)nrow[0] * 64 + gq * 16);
#else
  {
    const uint4* src = (const uint4*)(h1bf + (size_t)nrow[0] * 64 + gq * 16);
    uint4* dst = (uint4*)(&Alds[0][ldoff]);
    dst[0] = src[0];
    dst[1] = src[1];
  }
#endif

  int cur = 0;
  for (int k = 0; k < KNBR; ++k) {
#if USE_ASYNC_LDS
    async_wait0();               // my async writes into Alds[cur] are done
#endif
    __syncthreads();             // everyone's tile-k data visible

    // prefetch tile k+1 into the other buffer (overlaps the WMMAs below)
    if (k + 1 < KNBR) {
#if USE_ASYNC_LDS
      async_copy32(&Alds[cur ^ 1][ldoff],
                   h1bf + (size_t)nrow[k + 1] * 64 + gq * 16);
#else
      const uint4* src = (const uint4*)(h1bf + (size_t)nrow[k + 1] * 64 + gq * 16);
      uint4* dst = (uint4*)(&Alds[cur ^ 1][ldoff]);
      dst[0] = src[0];
      dst[1] = src[1];
#endif
    }

    const unsigned short* abuf = &Alds[cur][0];
    #pragma unroll
    for (int half = 0; half < 2; ++half) {
      const unsigned short* abase = abuf + arow * ROWS + half * 32 + ahi;
      v8u av;
      #pragma unroll
      for (int j = 0; j < 8; ++j) {
        int off = ((j & 4) ? 16 : 0) + ((j & 3) << 1);
        av[j] = *(const unsigned int*)(abase + off);
      }
      v16bf afrag = __builtin_bit_cast(v16bf, av);

      const int kb = k * 2 + half;
      const unsigned short* bbase =
          w2swz + (((size_t)kb * 16 + w_n * 8) * 32 + lane) * 16;
      #pragma unroll
      for (int t = 0; t < 8; ++t) {
        v8u bv = *(const v8u*)(bbase + t * 512);
        v16bf bfrag = __builtin_bit_cast(v16bf, bv);
        acc[t] = __builtin_amdgcn_wmma_f32_16x16x32_bf16(
            false, afrag, false, bfrag, (short)0, acc[t], false, false);
      }
    }
    cur ^= 1;
  }

  // ---- fused heads: per-voxel s = relu(c+b2).score_w, g = relu(c+b2).sigma_w
  float b2v[8], swv[8], gwv[8];
  #pragma unroll
  for (int t = 0; t < 8; ++t) {
    int col = w_n * 128 + t * 16 + (lane & 15);
    b2v[t] = b2[col];
    swv[t] = score_w[col];
    gwv[t] = sigma_w[col];
  }
  #pragma unroll
  for (int i = 0; i < 8; ++i) {
    float s = 0.f, g = 0.f;
    #pragma unroll
    for (int t = 0; t < 8; ++t) {
      float v = fmaxf(acc[t][i] + b2v[t], 0.0f);
      s = fmaf(v, swv[t], s);
      g = fmaf(v, gwv[t], g);
    }
    // reduce over the 16 lanes (columns) sharing this row
    #pragma unroll
    for (int off = 8; off >= 1; off >>= 1) {
      s += __shfl_xor(s, off, 32);
      g += __shfl_xor(g, off, 32);
    }
    if ((lane & 15) == 0) {
      int row = w_m * 16 + i + ((lane >> 4) << 3);  // C layout: M = i + 8*(lane/16)
      red_s[w_n * 64 + row] = s;
      red_g[w_n * 64 + row] = g;
    }
  }
  __syncthreads();
  if (tid < 64) {
    s_vox[m0 + tid] = red_s[tid] + red_s[64 + tid];
    g_vox[m0 + tid] = red_g[tid] + red_g[64 + tid];
  }
}

// ----------------------- per-point heads + softplus ------------------------
__global__ void sn_heads_kernel(const float* __restrict__ s_vox,
                                const float* __restrict__ g_vox,
                                const int*   __restrict__ q2o,
                                const float* __restrict__ score_b,
                                const float* __restrict__ sigma_b,
                                float* __restrict__ out, int np) {
  int p = blockIdx.x * blockDim.x + threadIdx.x;
  if (p >= np) return;
  int v = q2o[p];
  out[p] = s_vox[v] + score_b[0];
  float x = g_vox[v] + sigma_b[0];
  out[np + p] = (x > 20.0f) ? x : log1pf(expf(x));
}

// ---------------------------------------------------------------------------
extern "C" void kernel_launch(void* const* d_in, const int* in_sizes, int n_in,
                              void* d_out, int out_size, void* d_ws, size_t ws_size,
                              hipStream_t stream) {
  (void)n_in; (void)out_size; (void)ws_size;
  const float* feats   = (const float*)d_in[0];
  const int*   nidx1   = (const int*)  d_in[1];
  const int*   nidx2   = (const int*)  d_in[2];
  const int*   q2o     = (const int*)  d_in[3];
  const float* W1      = (const float*)d_in[4];
  const float* b1      = (const float*)d_in[5];
  const float* W2      = (const float*)d_in[6];
  const float* b2      = (const float*)d_in[7];
  const float* score_w = (const float*)d_in[8];
  const float* score_b = (const float*)d_in[9];
  const float* sigma_w = (const float*)d_in[10];
  const float* sigma_b = (const float*)d_in[11];
  float* out = (float*)d_out;
  const int np = in_sizes[3];  // B*P = 100000

  // workspace layout
  char* ws = (char*)d_ws;
  unsigned short* h1bf  = (unsigned short*)ws;                       // N*64 bf16
  size_t off = (size_t)NVOX * 64 * sizeof(unsigned short);           // 25.6 MB
  unsigned short* w2swz = (unsigned short*)(ws + off);
  off += (size_t)KTOT * COUT * sizeof(unsigned short);               // 0.88 MB
  float* s_vox = (float*)(ws + off);
  off += (size_t)NVOX * sizeof(float);
  float* g_vox = (float*)(ws + off);

  sn_conv1_kernel<<<NVOX / 4, 256, 0, stream>>>(feats, nidx1, W1, b1, h1bf);
  sn_w2swz_kernel<<<(KTOT * COUT + 255) / 256, 256, 0, stream>>>(W2, w2swz);
  sn_conv2_gemm_kernel<<<NVOX / 64, 256, 0, stream>>>(
      h1bf, nidx2, w2swz, b2, score_w, sigma_w, s_vox, g_vox);
  sn_heads_kernel<<<(np + 255) / 256, 256, 0, stream>>>(
      s_vox, g_vox, q2o, score_b, sigma_b, out, np);
}
